// MPNNEncoder_2989297238495
// MI455X (gfx1250) — matrix operand order repaired and checked by reading, not compiled, not run on hardware
//
#include <hip/hip_runtime.h>
#include <stdint.h>

// ---------------------------------------------------------------------------
// MPNN encoder for MI455X (gfx1250, wave32, WMMA).
// Dominant GEMM: msg = (h ⊗ x) @ W2r  -> M=E, N=64, K=8192, bf16 WMMA.
// A-fragments built in registers (v_pk_mul_bf16); B prepacked (1MB) and
// staged block-cooperatively through double-buffered LDS (48KB) using
// global_load_async_to_lds_b128 (ASYNCcnt DMA, zero data VGPRs) so the
// copy overlaps the WMMAs of the current stage.
// ---------------------------------------------------------------------------

typedef __bf16 v16bf __attribute__((ext_vector_type(16)));
typedef float  v8f   __attribute__((ext_vector_type(8)));

#define DIV_UP(a, b) (((a) + (b) - 1) / (b))

static constexpr int D           = 64;
static constexpr int FIN         = 16;
static constexpr int HDIM        = 128;  // edge-MLP hidden width
static constexpr int NCHUNK      = 258;  // 256 GEMM K-chunks of 32 (K=8192) + 2 bias chunks
static constexpr int STAGE_CHUNKS= 6;    // chunks per LDS stage (even -> static parity)
static constexpr int NCHUNK_PAD  = 264;  // padded to 44 stages of 6 (chunks 258..263 zero)
static constexpr int NSTAGE      = NCHUNK_PAD / STAGE_CHUNKS;        // 44
static constexpr int STAGE_US    = STAGE_CHUNKS * 4 * 32 * 16;       // ushorts per stage (12288 = 24KB)
static constexpr int STAGE_BYTES = STAGE_US * 2;                     // 24576
static constexpr int B128_PER_THR= STAGE_BYTES / (256 * 16);         // 6 async b128 per thread/stage

// ---------------- helpers ----------------

__device__ __forceinline__ unsigned short f2bf(float f) {
    unsigned u = __float_as_uint(f);
    u += 0x7FFFu + ((u >> 16) & 1u);          // round-to-nearest-even
    return (unsigned short)(u >> 16);
}
__device__ __forceinline__ unsigned pack2bf(float lo, float hi) {
    return (unsigned)f2bf(lo) | ((unsigned)f2bf(hi) << 16);
}
__device__ __forceinline__ unsigned pk_mul_bf16(unsigned a, unsigned b) {
    unsigned d;
    asm("v_pk_mul_bf16 %0, %1, %2" : "=v"(d) : "v"(a), "v"(b));
    return d;
}
// Async DMA: global (SGPR base + per-lane 32-bit offset) -> LDS (per-lane byte addr).
// Tracked by ASYNCcnt; no data VGPRs.
__device__ __forceinline__ void async_ld_b128(unsigned lds_off, const unsigned short* sbase,
                                              unsigned goff) {
    asm volatile("global_load_async_to_lds_b128 %0, %1, %2"
                 :: "v"(lds_off), "v"(goff), "s"(sbase)
                 : "memory");
}
__device__ __forceinline__ void wait_async0() {
    asm volatile("s_wait_asynccnt 0x0" ::: "memory");
}
__device__ __forceinline__ void atomAddF(float* p, float v) {
    (void)__hip_atomic_fetch_add(p, v, __ATOMIC_RELAXED, __HIP_MEMORY_SCOPE_AGENT);
}
__device__ __forceinline__ void atomMaxF(float* addr, float val) {
    unsigned* ua = (unsigned*)addr;
    unsigned old = *ua;
    while (__uint_as_float(old) < val) {
        unsigned assumed = old;
        old = atomicCAS(ua, assumed, __float_as_uint(val));
        if (old == assumed) break;
    }
}
__device__ __forceinline__ float sigmoidf(float x) { return 1.0f / (1.0f + __expf(-x)); }

// ---------------- small utility kernels ----------------

__global__ void fill_f32_k(float* p, int n, float v) {
    int t = blockIdx.x * blockDim.x + threadIdx.x;
    if (t < n) p[t] = v;
}
__global__ void fill_i32_k(int* p, int n, int v) {
    int t = blockIdx.x * blockDim.x + threadIdx.x;
    if (t < n) p[t] = v;
}
__global__ void deg_k(const int* __restrict__ dst, int* __restrict__ deg, int E) {
    int e = blockIdx.x * blockDim.x + threadIdx.x;
    if (e < E) atomicAdd(&deg[dst[e]], 1);
}

// lin0 + ReLU: X[n,d] = relu(x[n,:16] . w[d,:16] + b[d])
__global__ void lin0_k(const float* __restrict__ x, const float* __restrict__ w,
                       const float* __restrict__ b, float* __restrict__ X, int N) {
    int t = blockIdx.x * blockDim.x + threadIdx.x;
    if (t >= N * D) return;
    int n = t >> 6, d = t & 63;
    const float* xr = x + (size_t)n * FIN;
    const float* wr = w + (size_t)d * FIN;
    float s = b[d];
#pragma unroll
    for (int j = 0; j < FIN; ++j) s += xr[j] * wr[j];
    X[t] = fmaxf(s, 0.0f);
}

// edge-MLP layer 1 + ReLU, output bf16: HID[e,j] = relu(ea[e,:5] . w1[j,:5] + b1[j])
__global__ void mlp1_k(const float* __restrict__ ea, const float* __restrict__ w1,
                       const float* __restrict__ b1, unsigned short* __restrict__ HID, int E) {
    int t = blockIdx.x * blockDim.x + threadIdx.x;
    if (t >= E * HDIM) return;
    int e = t >> 7, j = t & 127;
    const float* ar = ea + (size_t)e * 5;
    const float* wr = w1 + (size_t)j * 5;
    float s = b1[j];
#pragma unroll
    for (int q = 0; q < 5; ++q) s += ar[q] * wr[q];
    HID[t] = f2bf(fmaxf(s, 0.0f));
}

// Pack W2r (+ mlp_b2 as chunks 256/257, zeros for 258..263) in WMMA B-fragment order:
// fragment(c,t) lane L element i: K_local = (i<8?i:i+8) + (L<16?0:8), N = t*16 + (L&15).
// Chunk c<256 covers kappa = c*32..c*32+31 with k = c>>1 (constant), d = (c&1)*32 + K_local.
__global__ void pack_w2_k(const float* __restrict__ w2, const float* __restrict__ b2,
                          unsigned short* __restrict__ W2B) {
    int tid = blockIdx.x * blockDim.x + threadIdx.x;   // one thread = one lane-fragment
    if (tid >= NCHUNK_PAD * 4 * 32) return;
    int lane = tid & 31;
    int t    = (tid >> 5) & 3;
    int c    = tid >> 7;
    int o     = t * 16 + (lane & 15);
    int kbase = (lane < 16) ? 0 : 8;
    unsigned short* outp = W2B + (size_t)tid * 16;
#pragma unroll
    for (int i = 0; i < 16; ++i) {
        int Kl = kbase + (i < 8 ? i : i + 8);
        float v = 0.0f;
        if (c < 256) {
            int k = c >> 1;
            int d = (c & 1) * 32 + Kl;
            v = w2[(size_t)(d * 64 + o) * HDIM + k];
        } else if (c < NCHUNK) {
            int d = (c - 256) * 32 + Kl;
            v = b2[d * 64 + o];
        }
        outp[i] = f2bf(v);
    }
}

// ---------------- the WMMA message kernel ----------------
// Per wave: 16 edges, msg tile 16x64 = 4 accumulators of v8f.
// A(16x32 bf16) element i at lane L: M = L&15, K_local = (i<8?i:i+8) + (L<16?0:8).
// For chunk c: k = c>>1 constant  =>  A = h[M][k] * x[M][(c&1)*32 + K_local]  (rank-1).
// B staged block-wide through double-buffered LDS via async-to-LDS DMA.
__global__ __launch_bounds__(256) void msg_wmma_k(
        const float* __restrict__ X, const unsigned short* __restrict__ HID,
        const unsigned short* __restrict__ W2B, const int* __restrict__ src,
        const int* __restrict__ dst, float* __restrict__ AGG, int E) {
    const int tid  = threadIdx.x;
    const int lane = tid & 31;
    const int wave = tid >> 5;
    const int base = (blockIdx.x * 8 + wave) * 16;

    const int  half   = lane >> 4;
    const int  M      = lane & 15;
    const int  myEdge = base + M;
    const bool valid  = myEdge < E;
    const int  eSafe  = valid ? myEdge : 0;      // branchless clamp (cndmask)
    const int  srcIdx = src[eSafe];

    __shared__ __align__(16) unsigned short lbs[2][STAGE_US];   // 2 x 24KB

    // LDS byte offsets (flat LDS aperture => low 32 bits of generic ptr = LDS addr)
    const unsigned ldsBuf0 = (unsigned)(uintptr_t)(&lbs[0][0]);
    const unsigned ldsBuf1 = (unsigned)(uintptr_t)(&lbs[1][0]);

    __builtin_prefetch(W2B, 0, 3);

    // Pre-convert this lane's x-fragment halves (d in [p*32, p*32+32)) to packed bf16.
    unsigned xu[2][8];
    const float4* xr = reinterpret_cast<const float4*>(X + (size_t)srcIdx * D);
#pragma unroll
    for (int p = 0; p < 2; ++p) {
        int b0 = (p * 32 + half * 8) >> 2;       // elements i=0..7  -> d = p*32 + half*8 + i
        int b1 = (p * 32 + 16 + half * 8) >> 2;  // elements i=8..15 -> d = p*32+16+half*8+(i-8)
        float4 f0 = xr[b0], f1 = xr[b0 + 1], f2 = xr[b1], f3 = xr[b1 + 1];
        xu[p][0] = pack2bf(f0.x, f0.y); xu[p][1] = pack2bf(f0.z, f0.w);
        xu[p][2] = pack2bf(f1.x, f1.y); xu[p][3] = pack2bf(f1.z, f1.w);
        xu[p][4] = pack2bf(f2.x, f2.y); xu[p][5] = pack2bf(f2.z, f2.w);
        xu[p][6] = pack2bf(f3.x, f3.y); xu[p][7] = pack2bf(f3.z, f3.w);
    }

    // Stage 0: async DMA global -> LDS[0]
    {
        unsigned lo = ldsBuf0 + (unsigned)tid * 16;
        unsigned go = (unsigned)tid * 16;
#pragma unroll
        for (int q = 0; q < B128_PER_THR; ++q)
            async_ld_b128(lo + q * 4096, W2B, go + q * 4096);
    }
    wait_async0();
    __syncthreads();

    v8f acc0 = {}, acc1 = {}, acc2 = {}, acc3 = {};
    const unsigned short* hrow = HID + (size_t)eSafe * HDIM;
    unsigned hkpk = 0;

    for (int s = 0; s < NSTAGE; ++s) {
        // Kick off DMA for stage s+1 into the other buffer; it overlaps this
        // stage's WMMAs. Overwrite-safe: readers of that buffer (stage s-1)
        // finished before the barrier that ended stage s-1.
        if (s + 1 < NSTAGE) {
            unsigned lo = ((s + 1) & 1 ? ldsBuf1 : ldsBuf0) + (unsigned)tid * 16;
            unsigned go = (unsigned)(s + 1) * STAGE_BYTES + (unsigned)tid * 16;
#pragma unroll
            for (int q = 0; q < B128_PER_THR; ++q)
                async_ld_b128(lo + q * 4096, W2B, go + q * 4096);
        }

        const v16bf* lb = reinterpret_cast<const v16bf*>(&lbs[s & 1][0]);
#pragma unroll
        for (int cc = 0; cc < STAGE_CHUNKS; ++cc) {
            const int c = s * STAGE_CHUNKS + cc;
            const int p = cc & 1;                    // == c&1 (stage size even)
            if (c >= 256) {
                hkpk = 0x3F803F80u;                  // bias/pad chunks: h == 1.0 (bf16)
            } else if (p == 0) {
                unsigned short hb = hrow[c >> 1];
                if (!valid) hb = 0;
                hkpk = ((unsigned)hb << 16) | (unsigned)hb;
            }
            union { unsigned u[8]; v16bf v; } au;
#pragma unroll
            for (int j = 0; j < 8; ++j) au.u[j] = pk_mul_bf16(hkpk, xu[p][j]);

            const v16bf* bp = lb + ((cc * 4) * 32 + lane);
            acc0 = __builtin_amdgcn_wmma_f32_16x16x32_bf16(false, au.v, false, bp[0],  (short)0, acc0, false, false);
            acc1 = __builtin_amdgcn_wmma_f32_16x16x32_bf16(false, au.v, false, bp[32], (short)0, acc1, false, false);
            acc2 = __builtin_amdgcn_wmma_f32_16x16x32_bf16(false, au.v, false, bp[64], (short)0, acc2, false, false);
            acc3 = __builtin_amdgcn_wmma_f32_16x16x32_bf16(false, au.v, false, bp[96], (short)0, acc3, false, false);
        }

        if (s + 1 < NSTAGE) {
            wait_async0();                            // my DMA for stage s+1 landed
            __syncthreads();                          // everyone's landed; stage s reads done
        }
    }

    // C/D layout: VGPR r holds M = r + half*8, N = lane&15. Scatter-add to AGG[dst].
    const int col = lane & 15;
#pragma unroll
    for (int r = 0; r < 8; ++r) {
        int row = half * 8 + r;
        int e   = base + row;
        if (e < E) {
            int di = dst[e];
            float* o = AGG + (size_t)di * D + col;
            atomAddF(o + 0,  acc0[r]);
            atomAddF(o + 16, acc1[r]);
            atomAddF(o + 32, acc2[r]);
            atomAddF(o + 48, acc3[r]);
        }
    }
}

// ---------------- GRU update (one block of 64 threads per node) ----------------
__global__ __launch_bounds__(64) void gru_k(
        float* __restrict__ X, const float* __restrict__ AGG, const int* __restrict__ DEG,
        const float* __restrict__ conv_b,
        const float* __restrict__ w_ih, const float* __restrict__ w_hh,
        const float* __restrict__ b_ih, const float* __restrict__ b_hh, int N) {
    int n = blockIdx.x;
    int d = threadIdx.x;
    if (n >= N) return;
    __shared__ float sm[D], sh[D];
    float degf = fmaxf((float)DEG[n], 1.0f);
    float m = fmaxf(AGG[(size_t)n * D + d] / degf + conv_b[d], 0.0f);
    float h = X[(size_t)n * D + d];
    sm[d] = m; sh[d] = h;
    __syncthreads();
    float ir = b_ih[d], iz = b_ih[D + d], in_ = b_ih[2 * D + d];
    float hr = b_hh[d], hz = b_hh[D + d], hn  = b_hh[2 * D + d];
    const float* wir = w_ih + (size_t)d * D;
    const float* wiz = w_ih + (size_t)(D + d) * D;
    const float* win = w_ih + (size_t)(2 * D + d) * D;
    const float* whr = w_hh + (size_t)d * D;
    const float* whz = w_hh + (size_t)(D + d) * D;
    const float* whn = w_hh + (size_t)(2 * D + d) * D;
    for (int j = 0; j < D; ++j) {
        float mj = sm[j], hj = sh[j];
        ir += mj * wir[j];  iz += mj * wiz[j];  in_ += mj * win[j];
        hr += hj * whr[j];  hz += hj * whz[j];  hn  += hj * whn[j];
    }
    float r  = sigmoidf(ir + hr);
    float z  = sigmoidf(iz + hz);
    float nn = tanhf(in_ + r * hn);
    X[(size_t)n * D + d] = (1.0f - z) * nn + z * h;
}

// ---------------- Set2Set ----------------

// LSTM cell (one block of 256 threads per batch element); also resets EMAX/ASUM/R for step.
__global__ __launch_bounds__(256) void lstm_k(
        const float* __restrict__ QS, float* __restrict__ HH, float* __restrict__ CC,
        const float* __restrict__ w_ih, const float* __restrict__ w_hh,
        const float* __restrict__ b_ih, const float* __restrict__ b_hh,
        float* __restrict__ EMAX, float* __restrict__ ASUM, float* __restrict__ R) {
    int b = blockIdx.x;
    int j = threadIdx.x;
    __shared__ float sq[2 * D], sh[D], sg[4 * D];
    if (j < 2 * D) sq[j] = QS[(size_t)b * 2 * D + j];
    if (j < D)     sh[j] = HH[(size_t)b * D + j];
    if (j == 0) { EMAX[b] = -3.402823466e38f; ASUM[b] = 0.0f; }
    if (j >= D && j < 2 * D) R[(size_t)b * D + (j - D)] = 0.0f;
    __syncthreads();
    float g = b_ih[j] + b_hh[j];
    const float* wi = w_ih + (size_t)j * (2 * D);
    const float* wh = w_hh + (size_t)j * D;
    for (int q = 0; q < 2 * D; ++q) g += sq[q] * wi[q];
    for (int q = 0; q < D; ++q)     g += sh[q] * wh[q];
    sg[j] = g;
    __syncthreads();
    if (j < D) {
        float i  = sg[j], f = sg[D + j], gg = sg[2 * D + j], o = sg[3 * D + j];
        float cn = sigmoidf(f) * CC[(size_t)b * D + j] + sigmoidf(i) * tanhf(gg);
        float hn = sigmoidf(o) * tanhf(cn);
        CC[(size_t)b * D + j] = cn;
        HH[(size_t)b * D + j] = hn;
    }
}

// e[n] = <X[n], q[batch[n]]>, plus segment max.
__global__ void attn_e_k(const float* __restrict__ X, const float* __restrict__ HH,
                         const int* __restrict__ batch, float* __restrict__ EV,
                         float* __restrict__ EMAX, int N) {
    int n = blockIdx.x * blockDim.x + threadIdx.x;
    if (n >= N) return;
    int b = batch[n];
    const float* xr = X + (size_t)n * D;
    const float* q  = HH + (size_t)b * D;
    float s = 0.0f;
#pragma unroll 8
    for (int d = 0; d < D; ++d) s += xr[d] * q[d];
    EV[n] = s;
    atomMaxF(&EMAX[b], s);
}

// a[n] = exp(e - emax), segment sum.
__global__ void attn_a_k(const float* __restrict__ EV, const float* __restrict__ EMAX,
                         const int* __restrict__ batch, float* __restrict__ AV,
                         float* __restrict__ ASUM, int N) {
    int n = blockIdx.x * blockDim.x + threadIdx.x;
    if (n >= N) return;
    int b = batch[n];
    float a = __expf(EV[n] - EMAX[b]);
    AV[n] = a;
    atomAddF(&ASUM[b], a);
}

// r[b] += (a/asum) * X[n]
__global__ void attn_r_k(const float* __restrict__ X, const float* __restrict__ AV,
                         const float* __restrict__ ASUM, const int* __restrict__ batch,
                         float* __restrict__ R, int N) {
    int t = blockIdx.x * blockDim.x + threadIdx.x;
    if (t >= N * D) return;
    int n = t >> 6, d = t & 63;
    int b = batch[n];
    float w = AV[n] / ASUM[b];
    atomAddF(&R[(size_t)b * D + d], w * X[t]);
}

__global__ void qstar_k(const float* __restrict__ HH, const float* __restrict__ R,
                        float* __restrict__ QS, int B_) {
    int t = blockIdx.x * blockDim.x + threadIdx.x;
    if (t >= B_ * 2 * D) return;
    int b = t >> 7, j = t & 127;
    QS[t] = (j < D) ? HH[(size_t)b * D + j] : R[(size_t)b * D + (j - D)];
}

__global__ void copy_out_k(const float* __restrict__ QS, float* __restrict__ out, int n) {
    int t = blockIdx.x * blockDim.x + threadIdx.x;
    if (t < n) out[t] = QS[t];
}

// ---------------------------------------------------------------------------

extern "C" void kernel_launch(void* const* d_in, const int* in_sizes, int n_in,
                              void* d_out, int out_size, void* d_ws, size_t ws_size,
                              hipStream_t stream) {
    (void)n_in; (void)ws_size;
    const float* x         = (const float*)d_in[0];
    const int*   eidx      = (const int*)d_in[1];
    const float* eattr     = (const float*)d_in[2];
    const int*   batch     = (const int*)d_in[3];
    const float* lin0_w    = (const float*)d_in[4];
    const float* lin0_b    = (const float*)d_in[5];
    const float* mlp_w1    = (const float*)d_in[6];
    const float* mlp_b1    = (const float*)d_in[7];
    const float* mlp_w2    = (const float*)d_in[8];
    const float* mlp_b2    = (const float*)d_in[9];
    const float* conv_b    = (const float*)d_in[10];
    const float* gru_w_ih  = (const float*)d_in[11];
    const float* gru_w_hh  = (const float*)d_in[12];
    const float* gru_b_ih  = (const float*)d_in[13];
    const float* gru_b_hh  = (const float*)d_in[14];
    const float* lstm_w_ih = (const float*)d_in[15];
    const float* lstm_w_hh = (const float*)d_in[16];
    const float* lstm_b_ih = (const float*)d_in[17];
    const float* lstm_b_hh = (const float*)d_in[18];

    const int N  = in_sizes[0] / FIN;
    const int E  = in_sizes[1] / 2;
    const int B_ = out_size / (2 * D);
    const int* src  = eidx;
    const int* dstp = eidx + E;

    // workspace carve-out (256B aligned)
    char* ws = (char*)d_ws;
    size_t off = 0;
    auto alloc = [&](size_t bytes) -> char* {
        char* p = ws + off;
        off = (off + bytes + 255) & ~(size_t)255;
        return p;
    };
    float*          X    = (float*)alloc((size_t)N * D * 4);
    float*          AGG  = (float*)alloc((size_t)N * D * 4);
    unsigned short* HID  = (unsigned short*)alloc((size_t)E * HDIM * 2);
    unsigned short* W2B  = (unsigned short*)alloc((size_t)NCHUNK_PAD * 4 * 32 * 16 * 2);
    int*            DEG  = (int*)alloc((size_t)N * 4);
    float*          EV   = (float*)alloc((size_t)N * 4);
    float*          AV   = (float*)alloc((size_t)N * 4);
    float*          EMAX = (float*)alloc((size_t)B_ * 4);
    float*          ASUM = (float*)alloc((size_t)B_ * 4);
    float*          R    = (float*)alloc((size_t)B_ * D * 4);
    float*          HH   = (float*)alloc((size_t)B_ * D * 4);
    float*          CC   = (float*)alloc((size_t)B_ * D * 4);
    float*          QS   = (float*)alloc((size_t)B_ * 2 * D * 4);

    // degree (scatter-mean denominator)
    fill_i32_k<<<DIV_UP(N, 256), 256, 0, stream>>>(DEG, N, 0);
    deg_k<<<DIV_UP(E, 256), 256, 0, stream>>>(dstp, DEG, E);
    // lin0 + relu
    lin0_k<<<DIV_UP(N * D, 256), 256, 0, stream>>>(x, lin0_w, lin0_b, X, N);
    // edge MLP layer 1 (bf16 out)
    mlp1_k<<<DIV_UP(E * HDIM, 256), 256, 0, stream>>>(eattr, mlp_w1, mlp_b1, HID, E);
    // pack W2r + bias (+zero pad) into WMMA B-fragment order (~1MB, L2-resident)
    pack_w2_k<<<DIV_UP(NCHUNK_PAD * 4 * 32, 256), 256, 0, stream>>>(mlp_w2, mlp_b2, W2B);

    const int tiles  = DIV_UP(E, 16);
    const int blocks = DIV_UP(tiles, 8);
    for (int it = 0; it < 3; ++it) {
        fill_f32_k<<<DIV_UP(N * D, 256), 256, 0, stream>>>(AGG, N * D, 0.0f);
        msg_wmma_k<<<blocks, 256, 0, stream>>>(X, HID, W2B, src, dstp, AGG, E);
        gru_k<<<N, 64, 0, stream>>>(X, AGG, DEG, conv_b,
                                    gru_w_ih, gru_w_hh, gru_b_ih, gru_b_hh, N);
    }

    // Set2Set (3 processing steps)
    fill_f32_k<<<DIV_UP(B_ * 2 * D, 256), 256, 0, stream>>>(QS, B_ * 2 * D, 0.0f);
    fill_f32_k<<<DIV_UP(B_ * D, 256), 256, 0, stream>>>(HH, B_ * D, 0.0f);
    fill_f32_k<<<DIV_UP(B_ * D, 256), 256, 0, stream>>>(CC, B_ * D, 0.0f);
    for (int s = 0; s < 3; ++s) {
        lstm_k<<<B_, 256, 0, stream>>>(QS, HH, CC, lstm_w_ih, lstm_w_hh,
                                       lstm_b_ih, lstm_b_hh, EMAX, ASUM, R);
        attn_e_k<<<DIV_UP(N, 256), 256, 0, stream>>>(X, HH, batch, EV, EMAX, N);
        attn_a_k<<<DIV_UP(N, 256), 256, 0, stream>>>(EV, EMAX, batch, AV, ASUM, N);
        attn_r_k<<<DIV_UP(N * D, 256), 256, 0, stream>>>(X, AV, ASUM, batch, R, N);
        qstar_k<<<DIV_UP(B_ * 2 * D, 256), 256, 0, stream>>>(HH, R, QS, B_);
    }
    copy_out_k<<<DIV_UP(B_ * 2 * D, 256), 256, 0, stream>>>(QS, (float*)d_out, B_ * 2 * D);
}
